// Critic_60997125538137
// MI455X (gfx1250) — matrix operand order repaired and checked by reading, not compiled, not run on hardware
//
#include <hip/hip_runtime.h>
#include <hip/hip_bf16.h>

typedef __attribute__((ext_vector_type(16))) _Float16 v16h;
typedef __attribute__((ext_vector_type(8)))  float    v8f;

#define BB   64
#define TT   1000
#define SS   96
#define AA   32
#define INPD 128
#define HH   256

struct BranchPtrs {
  const _Float16* fc1p;   // packed fc1 weight  [4 kc][16 nt] blocks
  const _Float16* wihp;   // packed W_ih        [8 kc][16 nt]
  const _Float16* whhp;   // packed W_hh        [8 kc][16 nt]
  const float*    fc1b;   // [256]
  const float*    bias2;  // b_ih + b_hh [256]
  const float*    fc2w;   // [256]
  const float*    fc2b;   // [1]
  _Float16*       U;      // [T][64 tiles][32 lanes][8] f16 (D-fragment order)
};

// ---- fragment loaders -------------------------------------------------------
// 16-bit A matrix 16x32 (MxK): lane L holds row m=L%16, kg=(L/16)*8.
// VGPRs 0..3 hold halves [kg, kg+8); VGPRs 4..7 hold halves [16+kg, 16+kg+8).
// Both groups are contiguous 16-byte runs -> two b128 loads.
__device__ __forceinline__ v16h frag_a(const _Float16* base, int ld, int row0,
                                       int k0, int lane) {
  int m  = row0 + (lane & 15);
  int kg = (lane >> 4) << 3;
  const _Float16* p = base + (size_t)m * ld + k0 + kg;
  union { v16h v; uint4 q[2]; } r;
  r.q[0] = *(const uint4*)(p);
  r.q[1] = *(const uint4*)(p + 16);
  return r.v;
}

// B fragments pre-packed: block (kc,nt) = 32 lanes x 16 halves contiguous.
__device__ __forceinline__ v16h frag_b(const _Float16* packed, int blk, int lane) {
  return *(const v16h*)(packed + ((size_t)blk * 32 + lane) * 16);
}

__device__ __forceinline__ v8f wmma_f16(v16h a, v16h b, v8f c) {
  return __builtin_amdgcn_wmma_f32_16x16x32_f16(false, a, false, b, (short)0, c,
                                                false, false);
}

// ---- weight packing: W[K][N] f32 row-major -> fragment blocks ---------------
__global__ void pack_b_kernel(const float* __restrict__ W, _Float16* __restrict__ out,
                              int K, int N) {
  int blk  = blockIdx.x;         // kc * (N/16) + nt
  int NT   = N >> 4;
  int kc   = blk / NT, nt = blk % NT;
  int lane = threadIdx.x;        // 0..31
  int kg   = (lane >> 4) * 8;
  int n    = nt * 16 + (lane & 15);
  union { _Float16 h[16]; float4 f4[2]; } r;
#pragma unroll
  for (int v = 0; v < 8; ++v) {
    int k          = kc * 32 + kg + 2 * (v & 3) + (v >> 2) * 16;
    r.h[2 * v]     = (_Float16)W[(size_t)k * N + n];
    r.h[2 * v + 1] = (_Float16)W[(size_t)(k + 1) * N + n];
  }
  float4* dst = (float4*)(out + ((size_t)blk * 32 + lane) * 16);
  dst[0] = r.f4[0];
  dst[1] = r.f4[1];
}

__global__ void bias_kernel(const float* bhh1, const float* bih1,
                            const float* bhh2, const float* bih2,
                            float* o1, float* o2) {
  int i = threadIdx.x;
  o1[i] = bhh1[i] + bih1[i];
  o2[i] = bhh2[i] + bih2[i];
}

// ---- fused input projection: U_t = relu(X_t@fc1 + b1)@W_ih + bias2 ----------
// One block per timestep t; M-dim = 64 batch rows, so GEMM2's D-fragments map
// 1:1 onto the scan's additive tiles -> U stored in fragment order.
__global__ __launch_bounds__(256) void proj_kernel(
    const float* __restrict__ state, const float* __restrict__ action,
    BranchPtrs p0, BranchPtrs p1) {
  BranchPtrs P = blockIdx.y ? p1 : p0;
  int t = blockIdx.x;

  __shared__ _Float16 xa[64][INPD];  // 16 KB
  __shared__ _Float16 x1[64][HH];    // 32 KB

  int tid = threadIdx.x;
  for (int i = tid; i < 64 * INPD; i += 256) {
    int b = i >> 7, c = i & 127;
    float v = (c < SS) ? state[((size_t)b * TT + t) * SS + c]
                       : action[((size_t)b * TT + t) * AA + (c - SS)];
    xa[b][c] = (_Float16)v;
  }
  __syncthreads();

  int wave = tid >> 5, lane = tid & 31;
  int nlo = lane & 15;
  int mhi = (lane >> 4) << 3;

  // GEMM1: [64x128]@[128x256] + b1, relu -> x1 (f16 LDS). 64 tiles / 8 waves.
#pragma unroll
  for (int tt = 0; tt < 8; ++tt) {
    int tile = wave + tt * 8;
    int mi = tile >> 4, nj = tile & 15;
    float bz = P.fc1b[nj * 16 + nlo];
    v8f acc = {bz, bz, bz, bz, bz, bz, bz, bz};
#pragma unroll
    for (int kc = 0; kc < INPD / 32; ++kc) {
      v16h a = frag_a(&xa[0][0], INPD, mi * 16, kc * 32, lane);
      v16h b = frag_b(P.fc1p, kc * (HH / 16) + nj, lane);
      acc = wmma_f16(a, b, acc);
    }
    int n = nj * 16 + nlo;
    int mb = mi * 16 + mhi;
#pragma unroll
    for (int r = 0; r < 8; ++r) {
      float v = acc[r];
      x1[mb + r][n] = (_Float16)(v > 0.f ? v : 0.f);
    }
  }
  __syncthreads();

  // GEMM2: [64x256]@[256x256] + bias2 -> U in D-fragment order (b128 stores)
#pragma unroll
  for (int tt = 0; tt < 8; ++tt) {
    int tile = wave + tt * 8;
    int mi = tile >> 4, nj = tile & 15;
    float bz = P.bias2[nj * 16 + nlo];
    v8f acc = {bz, bz, bz, bz, bz, bz, bz, bz};
#pragma unroll
    for (int kc = 0; kc < HH / 32; ++kc) {
      v16h a = frag_a(&x1[0][0], HH, mi * 16, kc * 32, lane);
      v16h b = frag_b(P.wihp, kc * (HH / 16) + nj, lane);
      acc = wmma_f16(a, b, acc);
    }
    union { uint4 q; _Float16 hh[8]; } uu;
#pragma unroll
    for (int r = 0; r < 8; ++r) uu.hh[r] = (_Float16)acc[r];
    *(uint4*)(P.U + (((size_t)t * 64 + tile) * 32 + lane) * 8) = uu.q;
  }
}

// ---- sequential scan: h <- sigmoid(h@W_hh + U_t); q = h.fc2w + fc2b ---------
// One persistent workgroup per branch (16 waves). h lives in LDS; each wave's
// W_hh B-fragments (fixed nj == wave) live entirely in registers.
__global__ __launch_bounds__(512) void scan_kernel(BranchPtrs p0, BranchPtrs p1,
                                                   const float* __restrict__ hn,
                                                   float* __restrict__ out) {
  BranchPtrs P = blockIdx.x ? p1 : p0;
  out += (size_t)blockIdx.x * BB * TT;

  extern __shared__ char smem[];
  _Float16(*h)[HH] = (_Float16(*)[HH])(smem);        // 32 KB
  float* w2s = (float*)(smem + 32768);               // 1 KB

  int tid = threadIdx.x;
  for (int i = tid; i < BB * HH; i += 512)
    h[i >> 8][i & 255] = (_Float16)hn[i];
  for (int i = tid; i < HH; i += 512) w2s[i] = P.fc2w[i];
  __syncthreads();

  int wave = tid >> 5, lane = tid & 31;
  int nlo = lane & 15;
  int mhi = (lane >> 4) << 3;
  float q_bias = P.fc2b[0];

  // Register-resident B fragments: this wave always computes N-tile nj == wave.
  v16h bfrag[8];
#pragma unroll
  for (int kc = 0; kc < 8; ++kc)
    bfrag[kc] = frag_b(P.whhp, kc * 16 + wave, lane);

  for (int t = 0; t < TT; ++t) {
    const _Float16* u = P.U + (size_t)t * 64 * 256;   // 64 tiles * 256 halves
    {  // prefetch next step's U (32 KB = 64 B/thread) into cache
      int tn = (t + 1 < TT) ? t + 1 : t;
      __builtin_prefetch((const char*)(P.U + (size_t)tn * 64 * 256) + tid * 64, 0, 0);
    }
    v8f res[4];
#pragma unroll
    for (int tt = 0; tt < 4; ++tt) {
      int tile = tt * 16 + wave;                      // mi = tt, nj = wave
      v8f acc = {};
#pragma unroll
      for (int kc = 0; kc < HH / 32; ++kc) {
        v16h a = frag_a(&h[0][0], HH, tt * 16, kc * 32, lane);
        acc = wmma_f16(a, bfrag[kc], acc);
      }
      union { uint4 q; _Float16 hh[8]; } uu;
      uu.q = *(const uint4*)(u + ((size_t)tile * 32 + lane) * 8);
#pragma unroll
      for (int r = 0; r < 8; ++r) {
        float x = acc[r] + (float)uu.hh[r];
        acc[r] = 1.0f / (1.0f + __expf(-x));
      }
      res[tt] = acc;
    }
    __syncthreads();  // all reads of old h done
#pragma unroll
    for (int tt = 0; tt < 4; ++tt) {
      int n = wave * 16 + nlo;
      int mb = tt * 16 + mhi;
#pragma unroll
      for (int r = 0; r < 8; ++r) h[mb + r][n] = (_Float16)res[tt][r];
    }
    __syncthreads();  // new h visible
    if (tid < BB) {   // fused q projection (vectorized LDS reads)
      float s = q_bias;
      const uint4* hp = (const uint4*)&h[tid][0];
#pragma unroll 8
      for (int i = 0; i < 32; ++i) {
        union { uint4 q; _Float16 e[8]; } x;
        x.q = hp[i];
#pragma unroll
        for (int j = 0; j < 8; ++j) s += (float)x.e[j] * w2s[i * 8 + j];
      }
      out[(size_t)tid * TT + t] = s;
    }
  }
}

// ---- host launch ------------------------------------------------------------
extern "C" void kernel_launch(void* const* d_in, const int* in_sizes, int n_in,
                              void* d_out, int out_size, void* d_ws, size_t ws_size,
                              hipStream_t stream) {
  const float* state  = (const float*)d_in[0];
  const float* action = (const float*)d_in[1];
  const float* hn     = (const float*)d_in[2];
  const float* fc11_w = (const float*)d_in[3];
  const float* fc11_b = (const float*)d_in[4];
  const float* W_hh1  = (const float*)d_in[5];
  const float* W_ih1  = (const float*)d_in[6];
  const float* b_hh1  = (const float*)d_in[7];
  const float* b_ih1  = (const float*)d_in[8];
  const float* fc12_w = (const float*)d_in[9];
  const float* fc12_b = (const float*)d_in[10];
  const float* fc21_w = (const float*)d_in[11];
  const float* fc21_b = (const float*)d_in[12];
  const float* W_hh2  = (const float*)d_in[13];
  const float* W_ih2  = (const float*)d_in[14];
  const float* b_hh2  = (const float*)d_in[15];
  const float* b_ih2  = (const float*)d_in[16];
  const float* fc22_w = (const float*)d_in[17];
  const float* fc22_b = (const float*)d_in[18];

  char* ws = (char*)d_ws;
  const size_t SZ_FC1P = (size_t)INPD * HH * 2;  // 65536 B
  const size_t SZ_WP   = (size_t)HH * HH * 2;    // 131072 B
  const size_t SZ_U    = (size_t)TT * 64 * 32 * 8 * 2;

  size_t o = 0;
  _Float16* fc1p0 = (_Float16*)(ws + o); o += SZ_FC1P;
  _Float16* fc1p1 = (_Float16*)(ws + o); o += SZ_FC1P;
  _Float16* wihp0 = (_Float16*)(ws + o); o += SZ_WP;
  _Float16* wihp1 = (_Float16*)(ws + o); o += SZ_WP;
  _Float16* whhp0 = (_Float16*)(ws + o); o += SZ_WP;
  _Float16* whhp1 = (_Float16*)(ws + o); o += SZ_WP;
  float*    bias0 = (float*)(ws + o);    o += 1024;
  float*    bias1 = (float*)(ws + o);    o += 1024;
  _Float16* U0    = (_Float16*)(ws + o); o += SZ_U;
  _Float16* U1    = (_Float16*)(ws + o); o += SZ_U;

  // pack weights into WMMA fragment order
  pack_b_kernel<<<(INPD / 32) * (HH / 16), 32, 0, stream>>>(fc11_w, fc1p0, INPD, HH);
  pack_b_kernel<<<(INPD / 32) * (HH / 16), 32, 0, stream>>>(fc21_w, fc1p1, INPD, HH);
  pack_b_kernel<<<(HH / 32) * (HH / 16), 32, 0, stream>>>(W_ih1, wihp0, HH, HH);
  pack_b_kernel<<<(HH / 32) * (HH / 16), 32, 0, stream>>>(W_ih2, wihp1, HH, HH);
  pack_b_kernel<<<(HH / 32) * (HH / 16), 32, 0, stream>>>(W_hh1, whhp0, HH, HH);
  pack_b_kernel<<<(HH / 32) * (HH / 16), 32, 0, stream>>>(W_hh2, whhp1, HH, HH);
  bias_kernel<<<1, HH, 0, stream>>>(b_hh1, b_ih1, b_hh2, b_ih2, bias0, bias1);

  BranchPtrs P0 = {fc1p0, wihp0, whhp0, fc11_b, bias0, fc12_w, fc12_b, U0};
  BranchPtrs P1 = {fc1p1, wihp1, whhp1, fc21_b, bias1, fc22_w, fc22_b, U1};

  // big parallel GEMM chain: U_t = relu(X_t@fc1+b1)@W_ih + bias2
  proj_kernel<<<dim3(TT, 2), 256, 0, stream>>>(state, action, P0, P1);

  // persistent sequential scan, one workgroup per branch
  size_t smem = 32768 + 1024;
  scan_kernel<<<2, 512, smem, stream>>>(P0, P1, hn, (float*)d_out);
}
